// GraphNetwork_91293824843796
// MI455X (gfx1250) — compile-verified, matrix operand order checked by reading, not run on hardware
//
#include <hip/hip_runtime.h>
#include <hip/hip_bf16.h>
#include <math.h>

typedef __attribute__((ext_vector_type(16))) _Float16 v16h;
typedef __attribute__((ext_vector_type(8)))  float    v8f;

#define Hdim 64
#define OUTD 16

// ---------------------------------------------------------------------------
// Kernel 1: collapse the fully-linear MLP chain into (W_eff, b_eff).
// grid.x = 2 (block 0: node MLP din=31, block 1: edge MLP din=7).
// Writes W_eff^T pre-swizzled into the WMMA B-fragment lane layout (f16),
// plus the f32 effective bias.
// ---------------------------------------------------------------------------
__global__ __launch_bounds__(1024)
void collapse_mlp_kernel(const float* __restrict__ nWin,  const float* __restrict__ nbin,
                         const float* __restrict__ nWhid, const float* __restrict__ nbhid,
                         const float* __restrict__ nWout, const float* __restrict__ nbout,
                         const float* __restrict__ eWin,  const float* __restrict__ ebin,
                         const float* __restrict__ eWhid, const float* __restrict__ ebhid,
                         const float* __restrict__ eWout, const float* __restrict__ ebout,
                         _Float16* __restrict__ nFrag, float* __restrict__ nBias,
                         _Float16* __restrict__ eFrag, float* __restrict__ eBias)
{
    const bool isNode = (blockIdx.x == 0);
    const float* Win  = isNode ? nWin  : eWin;
    const float* bin  = isNode ? nbin  : ebin;
    const float* Whid = isNode ? nWhid : eWhid;
    const float* bhid = isNode ? nbhid : ebhid;
    const float* Wout = isNode ? nWout : eWout;
    const float* bout = isNode ? nbout : ebout;
    _Float16* wfrag   = isNode ? nFrag : eFrag;
    float*    beff    = isNode ? nBias : eBias;
    const int din     = isNode ? 31 : 7;

    __shared__ float bufA[Hdim * OUTD];
    __shared__ float bufB[Hdim * OUTD];
    __shared__ float bvA[Hdim];
    __shared__ float bvB[Hdim];

    const int t = threadIdx.x;

    // P = Wout (64x16)
    for (int i = t; i < Hdim * OUTD; i += blockDim.x) bufA[i] = Wout[i];
    if (t < Hdim) bvA[t] = bin[t];
    __syncthreads();

    // Right-collapse the 64x64 hidden chain: P <- Whid[layer] @ P
    float* src = bufA;
    float* dst = bufB;
    for (int layer = Hdim - 1; layer >= 0; --layer) {
        const float* W = Whid + (size_t)layer * Hdim * Hdim;
        for (int i = t; i < Hdim * OUTD; i += blockDim.x) {
            const int r = i >> 4, c = i & 15;
            float acc = 0.f;
#pragma unroll 8
            for (int k = 0; k < Hdim; ++k) acc += W[r * Hdim + k] * src[k * OUTD + c];
            dst[i] = acc;
        }
        __syncthreads();
        float* tmp = src; src = dst; dst = tmp;
    }

    // Left-collapse the bias: b <- b @ Whid[layer] + bhid[layer]
    float* bs = bvA;
    float* bd = bvB;
    for (int layer = 0; layer < Hdim; ++layer) {
        const float* W = Whid + (size_t)layer * Hdim * Hdim;
        if (t < Hdim) {
            float acc = bhid[(size_t)layer * Hdim + t];
#pragma unroll 8
            for (int k = 0; k < Hdim; ++k) acc += bs[k] * W[k * Hdim + t];
            bd[t] = acc;
        }
        __syncthreads();
        float* tmp = bs; bs = bd; bd = tmp;
    }

    // W_eff = Win @ P  (din x 16, zero-padded to 32 x 16), swizzled into the
    // v_wmma_f32_16x16x32_f16 B-matrix fragment layout:
    //   element (k, n) -> lane = n + 16*((k>>3)&1), half = (k&7) + 8*(k>>4)
    for (int i = t; i < 32 * OUTD; i += blockDim.x) {
        const int k = i >> 4, n = i & 15;
        float acc = 0.f;
        if (k < din) {
#pragma unroll 8
            for (int j = 0; j < Hdim; ++j) acc += Win[k * Hdim + j] * src[j * OUTD + n];
        }
        const int lane = n + (((k >> 3) & 1) << 4);
        const int half = (k & 7) + ((k >> 4) << 3);
        wfrag[lane * 16 + half] = (_Float16)acc;
    }
    // b_eff = b @ Wout + bout
    if (t < OUTD) {
        float acc = bout[t];
#pragma unroll 8
        for (int k = 0; k < Hdim; ++k) acc += bs[k] * Wout[k * OUTD + t];
        beff[t] = acc;
    }
}

// ---------------------------------------------------------------------------
// Store helper: D-matrix f32 layout is lane = n + 16*(m/8), vgpr = m%8.
// For each accumulator VGPR, lanes 0-15 / 16-31 each cover one contiguous
// 64B output row segment -> a clause of coalesced b32 stores. row0 is
// scalar (readfirstlane'd) so the fast/tail split is a scalar branch.
// ---------------------------------------------------------------------------
__device__ __forceinline__
void store_tile(float* __restrict__ out, const v8f& c,
                int row0, int grp, int n, int R)
{
    if (row0 + 16 <= R) {          // scalar fast path: no per-lane guards
#pragma unroll
        for (int v = 0; v < 8; ++v) {
            const int orow = row0 + (grp << 3) + v;
            out[(size_t)orow * OUTD + n] = c[v];
        }
    } else {                       // tail tile (never taken for these shapes)
#pragma unroll
        for (int v = 0; v < 8; ++v) {
            const int orow = row0 + (grp << 3) + v;
            if (orow < R) out[(size_t)orow * OUTD + n] = c[v];
        }
    }
}

// ---------------------------------------------------------------------------
// Kernel 2: latent_nodes = [nodes | g] @ W_eff + b_eff via one WMMA per
// 16-row tile. Bias is folded into the WMMA C operand (C(m,n) = b_eff[n]),
// so the accumulators are stored directly. One wave32 per tile.
// ---------------------------------------------------------------------------
__global__ __launch_bounds__(256)
void apply_nodes_kernel(const float* __restrict__ nodes,
                        const float* __restrict__ globals_,
                        const _Float16* __restrict__ wfrag,
                        const float* __restrict__ bias,
                        float* __restrict__ out,
                        int R, int N)
{
    const int lane = threadIdx.x & 31;
    const int wave = threadIdx.x >> 5;
    const int tile = blockIdx.x * (blockDim.x >> 5) + wave;
    // tile is wave-invariant: force it scalar so tile-level branches are SALU.
    const int row0 = __builtin_amdgcn_readfirstlane(tile * 16);
    if (row0 >= R) return;                 // scalar, wave-uniform exit

    const int m   = lane & 15;
    const int grp = lane >> 4;
    int row = row0 + m;
    if (row >= R) row = R - 1;             // only matters on (nonexistent) tail

    // Batch index is tile-uniform unless the tile straddles a batch boundary
    // (at most 1 tile in the whole grid) -> scalar division in the fast path.
    const int b0 = row0 / N;
    const int b1 = (row0 + 15 < R ? row0 + 15 : R - 1) / N;
    const float g = (b0 == b1) ? globals_[b0] : globals_[row / N];

    const float* __restrict__ x = nodes + (size_t)row * 30;

    // A fragment (16x32 f16): element (m,k) -> lane = m + 16*((k>>3)&1),
    // half = (k&7) + 8*(k>>4).
    v16h a;
#pragma unroll
    for (int h = 0; h < 16; ++h) {
        const int k = (h & 7) + (grp << 3) + ((h >> 3) << 4);
        float v = 0.f;
        if (k < 30)       v = x[k];
        else if (k == 30) v = g;
        a[h] = (_Float16)v;
    }

    const v16h bm = *(const v16h*)(wfrag + lane * 16);

    // C(m,n) = bias[n]: every element this lane owns has column n = lane&15.
    const int n = lane & 15;
    const float bn = bias[n];
    v8f c;
#pragma unroll
    for (int v = 0; v < 8; ++v) c[v] = bn;

    c = __builtin_amdgcn_wmma_f32_16x16x32_f16(false, a, false, bm,
                                               (short)0, c, false, false);

    store_tile(out, c, row0, grp, n, R);
}

// ---------------------------------------------------------------------------
// Kernel 3: edge features [e | d | ||d||] (K=7, zero-padded to 32) @ W_eff.
// Lanes 0-15 each build one row (gather sender/receiver positions); lanes
// 16-31 carry zero A fragments (their K-halves 8..15 / 24..31 are padding).
// ---------------------------------------------------------------------------
__global__ __launch_bounds__(256)
void apply_edges_kernel(const float* __restrict__ nodes,
                        const float* __restrict__ edges,
                        const int* __restrict__ senders,
                        const int* __restrict__ receivers,
                        const _Float16* __restrict__ wfrag,
                        const float* __restrict__ bias,
                        float* __restrict__ out,
                        int R, int E, int N)
{
    const int lane = threadIdx.x & 31;
    const int wave = threadIdx.x >> 5;
    const int tile = blockIdx.x * (blockDim.x >> 5) + wave;
    const int row0 = __builtin_amdgcn_readfirstlane(tile * 16);
    if (row0 >= R) return;                 // scalar, wave-uniform exit

    const int m   = lane & 15;
    const int grp = lane >> 4;

    v16h a;
#pragma unroll
    for (int h = 0; h < 16; ++h) a[h] = (_Float16)0.f;

    if (grp == 0) {                        // reconverges before the WMMA
        int row = row0 + m;
        if (row >= R) row = R - 1;
        // Tile-uniform batch index except on the single straddling tile.
        const int b0 = row0 / E;
        const int b1 = (row0 + 15 < R ? row0 + 15 : R - 1) / E;
        const int b  = (b0 == b1) ? b0 : (row / E);
        const float* __restrict__ nb = nodes + (size_t)b * N * 30;
        const int si = senders[row];
        const int ri = receivers[row];
        const float* ps = nb + (size_t)si * 30;
        const float* pr = nb + (size_t)ri * 30;
        const float dx = ps[0] - pr[0];
        const float dy = ps[1] - pr[1];
        const float dz = ps[2] - pr[2];
        const float nrm = sqrtf(dx * dx + dy * dy + dz * dz);
        const float* ef = edges + (size_t)row * 3;
        a[0] = (_Float16)ef[0];
        a[1] = (_Float16)ef[1];
        a[2] = (_Float16)ef[2];
        a[3] = (_Float16)dx;
        a[4] = (_Float16)dy;
        a[5] = (_Float16)dz;
        a[6] = (_Float16)nrm;
    }

    const v16h bm = *(const v16h*)(wfrag + lane * 16);

    const int n = lane & 15;
    const float bn = bias[n];
    v8f c;
#pragma unroll
    for (int v = 0; v < 8; ++v) c[v] = bn;

    c = __builtin_amdgcn_wmma_f32_16x16x32_f16(false, a, false, bm,
                                               (short)0, c, false, false);

    store_tile(out, c, row0, grp, n, R);
}

// ---------------------------------------------------------------------------
extern "C" void kernel_launch(void* const* d_in, const int* in_sizes, int n_in,
                              void* d_out, int out_size, void* d_ws, size_t ws_size,
                              hipStream_t stream) {
    const float* nodes     = (const float*)d_in[0];
    const float* globals_  = (const float*)d_in[1];
    const float* edges     = (const float*)d_in[2];
    const int*   senders   = (const int*)d_in[3];
    const int*   receivers = (const int*)d_in[4];
    const float* nWin  = (const float*)d_in[5];
    const float* nbin  = (const float*)d_in[6];
    const float* nWhid = (const float*)d_in[7];
    const float* nbhid = (const float*)d_in[8];
    const float* nWout = (const float*)d_in[9];
    const float* nbout = (const float*)d_in[10];
    const float* eWin  = (const float*)d_in[11];
    const float* ebin  = (const float*)d_in[12];
    const float* eWhid = (const float*)d_in[13];
    const float* ebhid = (const float*)d_in[14];
    const float* eWout = (const float*)d_in[15];
    const float* ebout = (const float*)d_in[16];

    const int B = in_sizes[1];                 // globals_ is (B,)
    const int N = in_sizes[0] / (B * 30);
    const int E = in_sizes[2] / (B * 3);

    float* out_nodes = (float*)d_out;
    float* out_edges = out_nodes + (size_t)B * N * OUTD;

    // Workspace layout (<5 KB): two 1 KB pre-swizzled f16 B fragments + biases.
    char* ws = (char*)d_ws;
    _Float16* nFrag = (_Float16*)(ws);
    _Float16* eFrag = (_Float16*)(ws + 1024);
    float*    nBias = (float*)(ws + 2048);
    float*    eBias = (float*)(ws + 2048 + 64);

    collapse_mlp_kernel<<<2, 1024, 0, stream>>>(
        nWin, nbin, nWhid, nbhid, nWout, nbout,
        eWin, ebin, eWhid, ebhid, eWout, ebout,
        nFrag, nBias, eFrag, eBias);

    const int Rn = B * N;
    const int Re = B * E;
    const int nodeTiles = (Rn + 15) / 16;
    const int edgeTiles = (Re + 15) / 16;

    apply_nodes_kernel<<<(nodeTiles + 7) / 8, 256, 0, stream>>>(
        nodes, globals_, nFrag, nBias, out_nodes, Rn, N);

    apply_edges_kernel<<<(edgeTiles + 7) / 8, 256, 0, stream>>>(
        nodes, edges, senders, receivers, eFrag, eBias, out_edges, Re, E, N);
}
